// DecoderRNNwithAttention_27642409517436
// MI455X (gfx1250) — compile-verified, matrix operand order checked by reading, not run on hardware
//
#include <hip/hip_runtime.h>
#include <hip/hip_bf16.h>

// ---------------------------------------------------------------------------
// Problem constants (from the reference)
// ---------------------------------------------------------------------------
constexpr int B    = 128;
constexpr int V    = 10000;
constexpr int Vp   = 10048;    // V padded to a multiple of 64 (GEMM N tiling)
constexpr int E    = 300;
constexpr int H    = 512;
constexpr int AD   = 512;      // attention dim
constexpr int ENC  = 2048;
constexpr int P    = 64;       // 8x8 spatial positions
constexpr int TCAP = 20;
constexpr int TL   = TCAP - 1; // 19 output timesteps
constexpr int NS   = TL - 1;   // 18 executed steps
constexpr int KIH  = E + ENC;          // 2348
constexpr int KCAT = E + ENC + H;      // 2860  ([emb, awe, h])
constexpr int KP   = 2880;             // KCAT padded to a multiple of 32

typedef __bf16 bf16_t;
typedef __attribute__((ext_vector_type(16))) __bf16 v16bf;
typedef __attribute__((ext_vector_type(8)))  float  v8f;

static __device__ inline bf16_t f2bf(float f) {
    union { float f; unsigned u; } v; v.f = f;
    unsigned r = v.u + 0x7FFFu + ((v.u >> 16) & 1u);   // round-to-nearest-even
    unsigned short h = (unsigned short)(r >> 16);
    return __builtin_bit_cast(bf16_t, h);
}
static __device__ inline float bf2f(bf16_t b) {
    unsigned short h = __builtin_bit_cast(unsigned short, b);
    union { unsigned u; float f; } v; v.u = ((unsigned)h) << 16;
    return v.f;
}
static __device__ inline float sigm(float x) { return 1.0f / (1.0f + __expf(-x)); }

// ---------------------------------------------------------------------------
// Wave-level bf16 WMMA GEMM:  out[M,N] = A[M,K] * W[N,K]^T + bias[N]
// One wave per block. Each wave computes a 32x64 tile: 2 A fragments x 4 B
// fragments -> 8 v_wmma per K-chunk, branch-free inner loop (accumulators
// stay pinned in VGPRs; all N/mask guards live in the store epilogue).
// Requirements: K % 32 == 0, M % 32 == 0, gridDim.x == Npad/64 where W has
// Npad >= N zero-padded rows; logical N (store bound) passed separately.
// Fragment layout per CDNA5 ISA 7.12.2 (16-bit A/B 16x32): lane half lh holds
// K = 8*lh + j (j<8) and K = 16 + 8*lh + (j-8) (j>=8).
// ---------------------------------------------------------------------------
static __device__ inline v16bf load_frag16(const bf16_t* __restrict__ row, int kb, int lh) {
    v16bf f;
#pragma unroll
    for (int j = 0; j < 8; ++j) f[j]     = row[kb +      8 * lh + j];
#pragma unroll
    for (int j = 0; j < 8; ++j) f[8 + j] = row[kb + 16 + 8 * lh + j];
    return f;
}

__global__ __launch_bounds__(32) void gemm_bf16_wmma(
    const bf16_t* __restrict__ Amat, const bf16_t* __restrict__ Wmat,
    const float* __restrict__ bias, float* __restrict__ out,
    int M, int N, int K, int ldo,
    const int* __restrict__ lengths, int step)
{
    const int lane = threadIdx.x & 31;
    const int lh = lane >> 4;     // lane half (0/1)
    const int lm = lane & 15;
    const int m0 = blockIdx.y * 32;
    const int n0 = blockIdx.x * 64;

    v8f acc[2][4];
#pragma unroll
    for (int u = 0; u < 2; ++u)
#pragma unroll
        for (int t = 0; t < 4; ++t) acc[u][t] = (v8f){};

    const bf16_t* Arow0 = Amat + (size_t)(m0 + lm) * K;
    const bf16_t* Arow1 = Amat + (size_t)(m0 + 16 + lm) * K;
    const bf16_t* Wrow  = Wmat + (size_t)(n0 + lm) * K;

    for (int kb = 0; kb < K; kb += 32) {
        v16bf a0 = load_frag16(Arow0, kb, lh);
        v16bf a1 = load_frag16(Arow1, kb, lh);
#pragma unroll
        for (int t = 0; t < 4; ++t) {
            v16bf b = load_frag16(Wrow + (size_t)(16 * t) * K, kb, lh);
            acc[0][t] = __builtin_amdgcn_wmma_f32_16x16x32_bf16(
                false, a0, false, b, (short)0, acc[0][t], false, false);
            acc[1][t] = __builtin_amdgcn_wmma_f32_16x16x32_bf16(
                false, a1, false, b, (short)0, acc[1][t], false, false);
        }
    }

    // Store epilogue (guards only here; inner loop above is branch-free)
#pragma unroll
    for (int t = 0; t < 4; ++t) {
        if (n0 + 16 * (t + 1) <= N) {              // uniform across the wave
            const int col = n0 + 16 * t + lm;
            const float bv = bias ? bias[col] : 0.0f;
#pragma unroll
            for (int u = 0; u < 2; ++u) {
#pragma unroll
                for (int r = 0; r < 8; ++r) {
                    const int row = m0 + 16 * u + r + 8 * lh;   // C: M = r + 8*lh
                    if (lengths == nullptr || lengths[row] > step)
                        out[(size_t)row * ldo + col] = acc[u][t][r] + bv;
                }
            }
        }
    }
}

// ---------------------------------------------------------------------------
// Elementwise / setup kernels
// ---------------------------------------------------------------------------
__global__ void zero_f32(float* __restrict__ p, int n) {
    int i = blockIdx.x * blockDim.x + threadIdx.x;
    if (i < n) p[i] = 0.0f;
}

__global__ void cvt_f2bf(const float* __restrict__ in, bf16_t* __restrict__ out, int n) {
    int i = blockIdx.x * blockDim.x + threadIdx.x;
    if (i < n) out[i] = f2bf(in[i]);
}

// converts n elements, writes zeros for indices in [n, ntot) (row padding)
__global__ void cvt_f2bf_pad(const float* __restrict__ in, bf16_t* __restrict__ out,
                             int n, int ntot) {
    int i = blockIdx.x * blockDim.x + threadIdx.x;
    if (i < ntot) out[i] = (i < n) ? f2bf(in[i]) : f2bf(0.0f);
}

// features [B, ENC, 8, 8] -> feats_bf [B, P, ENC] (bf16, row-major over ENC)
__global__ void feats_transpose(const float* __restrict__ features, bf16_t* __restrict__ feats_bf) {
    int i = blockIdx.x * blockDim.x + threadIdx.x;      // b*P*ENC + p*ENC + e
    if (i >= B * P * ENC) return;
    int e = i & (ENC - 1);
    int bp = i >> 11;            // ENC = 2048
    int p = bp & (P - 1);
    int b = bp >> 6;
    feats_bf[i] = f2bf(features[((size_t)b * ENC + e) * P + p]);
}

// init_bf[b][e] = mean over the 64 spatial positions (contiguous in features)
__global__ void init_mean(const float* __restrict__ features, bf16_t* __restrict__ init_bf) {
    int i = blockIdx.x * blockDim.x + threadIdx.x;      // b*ENC + e
    if (i >= B * ENC) return;
    const float* f = features + (size_t)i * P;
    float s = 0.0f;
#pragma unroll 8
    for (int p = 0; p < P; ++p) s += f[p];
    init_bf[i] = f2bf(s * (1.0f / (float)P));
}

// Wihh_bf[n][k] = [Wih | Whh | 0-pad] as bf16, k in [0, KP)
__global__ void pack_wihh(const float* __restrict__ Wih, const float* __restrict__ Whh,
                          bf16_t* __restrict__ Wp) {
    int i = blockIdx.x * blockDim.x + threadIdx.x;
    if (i >= 4 * H * KP) return;
    int n = i / KP;
    int k = i - n * KP;
    float v;
    if (k < KIH)       v = Wih[(size_t)n * KIH + k];
    else if (k < KCAT) v = Whh[(size_t)n * H + (k - KIH)];
    else               v = 0.0f;
    Wp[i] = f2bf(v);
}

__global__ void add_bias2(const float* __restrict__ a, const float* __restrict__ b,
                          float* __restrict__ o, int n) {
    int i = blockIdx.x * blockDim.x + threadIdx.x;
    if (i < n) o[i] = a[i] + b[i];
}

__global__ void emb_gather(const float* __restrict__ emb_W, const int* __restrict__ captions,
                           bf16_t* __restrict__ emb_bf) {
    int i = blockIdx.x * blockDim.x + threadIdx.x;      // (b*TCAP+t)*E + e
    if (i >= B * TCAP * E) return;
    int e = i % E;
    int bt = i / E;
    int tok = captions[bt];
    emb_bf[i] = f2bf(emb_W[(size_t)tok * E + e]);
}

// ---------------------------------------------------------------------------
// Per-step attention kernels (reduction shaped -> f32 VALU + wave32 shuffles)
// ---------------------------------------------------------------------------
// logits[b*P+p] = sum_a relu(enc_attn[b,p,a] + dec_attn[b,a]) * Wal[a] + bal
__global__ void attn_logits(const float* __restrict__ enc_attn, const float* __restrict__ dec_attn,
                            const float* __restrict__ Wal, const float* __restrict__ bal,
                            float* __restrict__ logits) {
    int idx = blockIdx.x * (blockDim.x >> 5) + (threadIdx.x >> 5);   // b*P + p
    int lane = threadIdx.x & 31;
    if (idx >= B * P) return;
    int b = idx >> 6;
    const float* e = enc_attn + (size_t)idx * AD;
    const float* d = dec_attn + (size_t)b * AD;
    float s = 0.0f;
    for (int a = lane; a < AD; a += 32) {
        float v = e[a] + d[a];
        v = v > 0.0f ? v : 0.0f;
        s += v * Wal[a];
    }
#pragma unroll
    for (int m = 16; m; m >>= 1) s += __shfl_xor(s, m, 32);
    if (lane == 0) logits[idx] = s + bal[0];
}

// softmax over P=64 per batch row; also writes the masked attention output
__global__ __launch_bounds__(32) void attn_softmax(
    const float* __restrict__ logits, float* __restrict__ aw,
    float* __restrict__ attn_out, const int* __restrict__ lengths, int step)
{
    int b = blockIdx.x;
    int lane = threadIdx.x;
    float v0 = logits[b * P + lane];
    float v1 = logits[b * P + 32 + lane];
    float mx = fmaxf(v0, v1);
#pragma unroll
    for (int m = 16; m; m >>= 1) mx = fmaxf(mx, __shfl_xor(mx, m, 32));
    float e0 = __expf(v0 - mx), e1 = __expf(v1 - mx);
    float s = e0 + e1;
#pragma unroll
    for (int m = 16; m; m >>= 1) s += __shfl_xor(s, m, 32);
    float inv = 1.0f / s;
    float a0 = e0 * inv, a1 = e1 * inv;
    aw[b * P + lane] = a0;
    aw[b * P + 32 + lane] = a1;
    if (lengths[b] > step) {
        float* o = attn_out + ((size_t)b * TL + step) * P;
        o[lane] = a0;
        o[lane + 32] = a1;
    }
}

// ctx[b,e] = sum_p feats[b,p,e] * aw[b,p]
__global__ void ctx_kernel(const bf16_t* __restrict__ feats_bf, const float* __restrict__ aw,
                           float* __restrict__ ctx) {
    int i = blockIdx.x * blockDim.x + threadIdx.x;      // b*ENC + e
    if (i >= B * ENC) return;
    int b = i >> 11;
    int e = i & (ENC - 1);
    const bf16_t* f = feats_bf + ((size_t)b * P) * ENC + e;
    const float* w = aw + b * P;
    float s = 0.0f;
#pragma unroll 4
    for (int p = 0; p < P; ++p) s += bf2f(f[(size_t)p * ENC]) * w[p];
    ctx[i] = s;
}

// xh[b,k] = [emb_i | sigmoid(gatelin)*ctx | h | 0-pad] in bf16, k in [0, KP)
__global__ void assemble_x(const bf16_t* __restrict__ emb_bf, const float* __restrict__ gatelin,
                           const float* __restrict__ ctxv, const float* __restrict__ h,
                           bf16_t* __restrict__ xh, int step) {
    int i = blockIdx.x * blockDim.x + threadIdx.x;
    if (i >= B * KP) return;
    int b = i / KP;
    int k = i - b * KP;
    bf16_t v;
    if (k < E) {
        v = emb_bf[((size_t)b * TCAP + step) * E + k];
    } else if (k < KIH) {
        int j = k - E;
        v = f2bf(sigm(gatelin[b * ENC + j]) * ctxv[b * ENC + j]);
    } else if (k < KCAT) {
        v = f2bf(h[b * H + (k - KIH)]);
    } else {
        v = f2bf(0.0f);
    }
    xh[i] = v;
}

// LSTM cell pointwise (torch gate order i,f,g,o); masked carry update
__global__ void lstm_pointwise(const float* __restrict__ g, float* __restrict__ h,
                               float* __restrict__ c, bf16_t* __restrict__ hnew_bf,
                               const int* __restrict__ lengths, int step) {
    int i = blockIdx.x * blockDim.x + threadIdx.x;      // b*H + j
    if (i >= B * H) return;
    int b = i >> 9;
    int j = i & (H - 1);
    const float* gb = g + (size_t)b * 4 * H;
    float ig = sigm(gb[j]);
    float fg = sigm(gb[H + j]);
    float gg = tanhf(gb[2 * H + j]);
    float og = sigm(gb[3 * H + j]);
    float cn = fg * c[i] + ig * gg;
    float hn = og * tanhf(cn);
    hnew_bf[i] = f2bf(hn);
    if (lengths[b] > step) { h[i] = hn; c[i] = cn; }
}

// ---------------------------------------------------------------------------
// Host launcher
// ---------------------------------------------------------------------------
static inline int ewg(long n, int bs = 256) { return (int)((n + bs - 1) / bs); }

extern "C" void kernel_launch(void* const* d_in, const int* in_sizes, int n_in,
                              void* d_out, int out_size, void* d_ws, size_t ws_size,
                              hipStream_t stream) {
    (void)in_sizes; (void)n_in; (void)out_size; (void)ws_size;

    const float* features = (const float*)d_in[0];
    const int*   captions = (const int*)d_in[1];
    const int*   lengths  = (const int*)d_in[2];
    const float* emb_W    = (const float*)d_in[3];
    const float* Wh0 = (const float*)d_in[4];  const float* bh0 = (const float*)d_in[5];
    const float* Wc0 = (const float*)d_in[6];  const float* bc0 = (const float*)d_in[7];
    const float* Wih = (const float*)d_in[8];  const float* bih = (const float*)d_in[9];
    const float* Whh = (const float*)d_in[10]; const float* bhh = (const float*)d_in[11];
    const float* Wea = (const float*)d_in[12]; const float* bea = (const float*)d_in[13];
    const float* Wda = (const float*)d_in[14]; const float* bda = (const float*)d_in[15];
    const float* Wal = (const float*)d_in[16]; const float* bal = (const float*)d_in[17];
    const float* Wb  = (const float*)d_in[18]; const float* bb  = (const float*)d_in[19];
    const float* Wout= (const float*)d_in[20]; const float* bout= (const float*)d_in[21];

    float* pred_out = (float*)d_out;                         // [B, TL, V]
    float* attn_out = pred_out + (size_t)B * TL * V;         // [B, TL, P]
    const long out_total = (long)B * TL * V + (long)B * TL * P;

    // ---- workspace carve (256B aligned) ----
    char* w = (char*)d_ws;
    size_t off = 0;
    auto carve = [&](size_t bytes) -> void* {
        void* p = w + off;
        off += (bytes + 255) & ~(size_t)255;
        return p;
    };
    bf16_t* feats_bf = (bf16_t*)carve((size_t)B * P * ENC * 2);
    float*  enc_attn = (float*) carve((size_t)B * P * AD * 4);
    bf16_t* emb_bf   = (bf16_t*)carve((size_t)B * TCAP * E * 2);
    bf16_t* Wh0_bf   = (bf16_t*)carve((size_t)H * ENC * 2);
    bf16_t* Wc0_bf   = (bf16_t*)carve((size_t)H * ENC * 2);
    bf16_t* Wea_bf   = (bf16_t*)carve((size_t)AD * ENC * 2);
    bf16_t* Wda_bf   = (bf16_t*)carve((size_t)AD * H * 2);
    bf16_t* Wb_bf    = (bf16_t*)carve((size_t)ENC * H * 2);
    bf16_t* Wout_bf  = (bf16_t*)carve((size_t)Vp * H * 2);   // zero-padded rows
    bf16_t* Wihh_bf  = (bf16_t*)carve((size_t)4 * H * KP * 2);
    float*  bihh     = (float*) carve((size_t)4 * H * 4);
    bf16_t* init_bf  = (bf16_t*)carve((size_t)B * ENC * 2);
    float*  hbuf     = (float*) carve((size_t)B * H * 4);
    float*  cbuf     = (float*) carve((size_t)B * H * 4);
    bf16_t* h_bf     = (bf16_t*)carve((size_t)B * H * 2);
    bf16_t* hnew_bf  = (bf16_t*)carve((size_t)B * H * 2);
    float*  dec_attn = (float*) carve((size_t)B * AD * 4);
    float*  logits   = (float*) carve((size_t)B * P * 4);
    float*  aw_buf   = (float*) carve((size_t)B * P * 4);
    float*  ctxv     = (float*) carve((size_t)B * ENC * 4);
    float*  gatelin  = (float*) carve((size_t)B * ENC * 4);
    bf16_t* xh_bf    = (bf16_t*)carve((size_t)B * KP * 2);
    float*  gbuf     = (float*) carve((size_t)B * 4 * H * 4);

    // ---- one-time setup ----
    zero_f32<<<ewg(out_total), 256, 0, stream>>>(pred_out, (int)out_total);

    cvt_f2bf<<<ewg((long)H * ENC), 256, 0, stream>>>(Wh0, Wh0_bf, H * ENC);
    cvt_f2bf<<<ewg((long)H * ENC), 256, 0, stream>>>(Wc0, Wc0_bf, H * ENC);
    cvt_f2bf<<<ewg((long)AD * ENC), 256, 0, stream>>>(Wea, Wea_bf, AD * ENC);
    cvt_f2bf<<<ewg((long)AD * H), 256, 0, stream>>>(Wda, Wda_bf, AD * H);
    cvt_f2bf<<<ewg((long)ENC * H), 256, 0, stream>>>(Wb, Wb_bf, ENC * H);
    cvt_f2bf_pad<<<ewg((long)Vp * H), 256, 0, stream>>>(Wout, Wout_bf, V * H, Vp * H);
    pack_wihh<<<ewg((long)4 * H * KP), 256, 0, stream>>>(Wih, Whh, Wihh_bf);
    add_bias2<<<ewg(4 * H), 256, 0, stream>>>(bih, bhh, bihh, 4 * H);
    emb_gather<<<ewg((long)B * TCAP * E), 256, 0, stream>>>(emb_W, captions, emb_bf);

    feats_transpose<<<ewg((long)B * P * ENC), 256, 0, stream>>>(features, feats_bf);
    init_mean<<<ewg((long)B * ENC), 256, 0, stream>>>(features, init_bf);

    // h0 / c0: [B,H] = init[B,ENC] x W[H,ENC]^T
    {
        dim3 g(H / 64, B / 32);
        gemm_bf16_wmma<<<g, 32, 0, stream>>>(init_bf, Wh0_bf, bh0, hbuf, B, H, ENC, H, nullptr, 0);
        gemm_bf16_wmma<<<g, 32, 0, stream>>>(init_bf, Wc0_bf, bc0, cbuf, B, H, ENC, H, nullptr, 0);
    }
    // enc_attn: [B*P, AD] = feats[B*P, ENC] x Wea[AD, ENC]^T
    {
        dim3 g(AD / 64, (B * P) / 32);
        gemm_bf16_wmma<<<g, 32, 0, stream>>>(feats_bf, Wea_bf, bea, enc_attn, B * P, AD, ENC, AD, nullptr, 0);
    }

    // ---- recurrent steps ----
    for (int step = 0; step < NS; ++step) {
        cvt_f2bf<<<ewg((long)B * H), 256, 0, stream>>>(hbuf, h_bf, B * H);

        // dec_attn = h @ Wda^T + bda   [B, AD]
        {
            dim3 g(AD / 64, B / 32);
            gemm_bf16_wmma<<<g, 32, 0, stream>>>(h_bf, Wda_bf, bda, dec_attn, B, AD, H, AD, nullptr, 0);
        }
        attn_logits<<<(B * P) / 8, 256, 0, stream>>>(enc_attn, dec_attn, Wal, bal, logits);
        attn_softmax<<<B, 32, 0, stream>>>(logits, aw_buf, attn_out, lengths, step);
        ctx_kernel<<<ewg((long)B * ENC), 256, 0, stream>>>(feats_bf, aw_buf, ctxv);

        // gate pre-activation = h @ Wb^T + bb   [B, ENC]
        {
            dim3 g(ENC / 64, B / 32);
            gemm_bf16_wmma<<<g, 32, 0, stream>>>(h_bf, Wb_bf, bb, gatelin, B, ENC, H, ENC, nullptr, 0);
        }
        assemble_x<<<ewg((long)B * KP), 256, 0, stream>>>(emb_bf, gatelin, ctxv, hbuf, xh_bf, step);

        // LSTM gates: g = [x|h] @ [Wih|Whh]^T + (bih+bhh)   [B, 4H], K = KP
        {
            dim3 g(4 * H / 64, B / 32);
            gemm_bf16_wmma<<<g, 32, 0, stream>>>(xh_bf, Wihh_bf, bihh, gbuf, B, 4 * H, KP, 4 * H, nullptr, 0);
        }
        lstm_pointwise<<<ewg((long)B * H), 256, 0, stream>>>(gbuf, hbuf, cbuf, hnew_bf, lengths, step);

        // preds = h_new @ Wout^T + bout, masked-store into pred_out[:, step, :]
        // (Wout_bf has Vp zero-padded rows so the inner loop is unguarded)
        {
            dim3 g(Vp / 64, B / 32);
            gemm_bf16_wmma<<<g, 32, 0, stream>>>(hnew_bf, Wout_bf, bout,
                                                 pred_out + (size_t)step * V,
                                                 B, V, H, TL * V, lengths, step);
        }
    }
}
